// NeuSum_4879082848347
// MI455X (gfx1250) — compile-verified, matrix operand order checked by reading, not run on hardware
//
#include <hip/hip_runtime.h>
#include <math.h>
#include <stdint.h>

// ---------------- constants from the reference ----------------
#define BB 128
#define LL 200
#define SH 512
#define D2 1024   // 2*SH
#define EH 512
#define AA 512
#define TT 3
#define G3 1536   // 3*EH

typedef __attribute__((ext_vector_type(16))) __bf16 v16bf;
typedef __attribute__((ext_vector_type(8)))  __bf16 v8bf;
typedef __attribute__((ext_vector_type(8)))  float  v8f;

__device__ __forceinline__ float sigf(float x) { return 1.0f / (1.0f + expf(-x)); }

// 16-byte async copy global -> LDS (ASYNCcnt-tracked, GV addressing).
// LDS byte address = low 32 bits of the generic pointer (FLAT aperture rule).
__device__ __forceinline__ void async_copy_b128(const void* g, void* l) {
    uint32_t lds_addr = (uint32_t)(uintptr_t)l;
    asm volatile("global_load_async_to_lds_b128 %0, %1, off"
                 :
                 : "v"(lds_addr), "v"(g)
                 : "memory");
}

// ---------------------------------------------------------------------------
// WMMA GEMM:  Y[M x N] = act( X[M x K] f32 (row stride ldx) @ Wb[N x K]^T + bias )
//   Wb is pre-converted bf16 (N x K row-major).
// Block = 128 threads (4 waves). 64x64 output tile; wave w owns rows [16w,16w+16),
// 4 accumulators cover the 64 columns. A: global f32 -> cvt -> LDS bf16.
// B: double-buffered global_load_async_to_lds_b128 of bf16 weight tiles.
// Requires: grid.y = M/64, grid.x = N/64, K % 32 == 0.
// ---------------------------------------------------------------------------
__global__ __launch_bounds__(128) void wmma_gemm(
    const float*  __restrict__ X, long long ldx,
    const __bf16* __restrict__ Wb,        // N x K bf16 row-major
    const float*  __restrict__ bias,      // N
    float* __restrict__ Y, int ldy,
    int K, int act)
{
    __shared__ __bf16 As[64][32];
    __shared__ __bf16 Bs[2][64][32];

    const int tileN = blockIdx.x * 64;
    const int tileM = blockIdx.y * 64;
    const int tid   = threadIdx.x;
    const int lane  = tid & 31;
    const int wv    = tid >> 5;
    const int mrow  = lane & 15;
    const int half  = lane >> 4;

    v8f zero = {};
    v8f acc[4];
#pragma unroll
    for (int s = 0; s < 4; ++s) acc[s] = zero;

    // A staging: thread -> (row, 16-wide k slice)
    const int lrow = tid >> 1;
    const int lk   = (tid & 1) << 4;

    // B async staging: 64x32 bf16 tile = 256 16B chunks; 2 chunks per thread.
    const int br0 = tid >> 2;            // chunk row (c = tid)
    const int br1 = (tid + 128) >> 2;    // chunk row (c = tid + 128)
    const int bsg = (tid & 3) << 3;      // k offset within tile (elements)

    // prologue: async-copy B tile 0 into buffer 0
    async_copy_b128(Wb + (size_t)(tileN + br0) * K + bsg, &Bs[0][br0][bsg]);
    async_copy_b128(Wb + (size_t)(tileN + br1) * K + bsg, &Bs[0][br1][bsg]);

    int cur = 0;
    for (int k0 = 0; k0 < K; k0 += 32) {
        // ---- issue async copy of the NEXT B tile into the other buffer ----
        const bool more = (k0 + 32 < K);
        if (more) {
            const int kn = k0 + 32;
            async_copy_b128(Wb + (size_t)(tileN + br0) * K + kn + bsg,
                            &Bs[cur ^ 1][br0][bsg]);
            async_copy_b128(Wb + (size_t)(tileN + br1) * K + kn + bsg,
                            &Bs[cur ^ 1][br1][bsg]);
        }

        // ---- stage A tile (64 x 32) f32 -> bf16 via registers ----
        {
            const float4* xp = (const float4*)(X + (size_t)(tileM + lrow) * ldx + k0 + lk);
#pragma unroll
            for (int u = 0; u < 4; ++u) {
                float4 f = xp[u];
                As[lrow][lk + 4*u + 0] = (__bf16)f.x;
                As[lrow][lk + 4*u + 1] = (__bf16)f.y;
                As[lrow][lk + 4*u + 2] = (__bf16)f.z;
                As[lrow][lk + 4*u + 3] = (__bf16)f.w;
            }
        }

        // wait for the CURRENT B tile's async copies (in-order completion):
        // 2 outstanding belong to the next tile when one was just issued.
        if (more) {
            asm volatile("s_wait_asynccnt 0x2" ::: "memory");
        } else {
            asm volatile("s_wait_asynccnt 0x0" ::: "memory");
        }
        __syncthreads();   // A ds_stores + all waves' async B data visible

        // ---- build A fragment (ISA 16-bit A layout: lane=M%16, K striped,
        //      upper half-wave +8 in K) ----
        v16bf afrag;
#pragma unroll
        for (int v = 0; v < 8; ++v) {
            const int kb = ((v >> 2) << 4) | (half << 3) | ((v & 3) << 1);
            afrag[2*v + 0] = As[16*wv + mrow][kb + 0];
            afrag[2*v + 1] = As[16*wv + mrow][kb + 1];
        }

#pragma unroll
        for (int s = 0; s < 4; ++s) {
            v16bf bfrag;
#pragma unroll
            for (int v = 0; v < 8; ++v) {
                const int kb = ((v >> 2) << 4) | (half << 3) | ((v & 3) << 1);
                bfrag[2*v + 0] = Bs[cur][16*s + mrow][kb + 0];
                bfrag[2*v + 1] = Bs[cur][16*s + mrow][kb + 1];
            }
            acc[s] = __builtin_amdgcn_wmma_f32_16x16x32_bf16(
                false, afrag, false, bfrag, (short)0, acc[s], false, false);
        }
        __syncthreads();   // protects As and Bs[cur^1] for the next iteration
        cur ^= 1;
    }

    // writeback: C/D layout — VGPR r: lanes 0-15 -> M=r, lanes 16-31 -> M=r+8
#pragma unroll
    for (int s = 0; s < 4; ++s) {
        const int col = tileN + 16*s + mrow;
        const float bv = bias[col];
#pragma unroll
        for (int r = 0; r < 8; ++r) {
            const int row = tileM + 16*wv + r + 8*half;
            float v = acc[s][r] + bv;
            if (act) v = tanhf(v);
            Y[(size_t)row * ldy + col] = v;
        }
    }
}

// ---------------------------------------------------------------------------
// f32 -> bf16 conversion (weights pre-pass); n must be a multiple of 8
// ---------------------------------------------------------------------------
__global__ void cvt_bf16_kernel(const float* __restrict__ s,
                                __bf16* __restrict__ d, int n)
{
    const int i = (blockIdx.x * blockDim.x + threadIdx.x) * 8;
    if (i >= n) return;
    const float4 a = *(const float4*)(s + i);
    const float4 b = *(const float4*)(s + i + 4);
    v8bf o;
    o[0] = (__bf16)a.x; o[1] = (__bf16)a.y; o[2] = (__bf16)a.z; o[3] = (__bf16)a.w;
    o[4] = (__bf16)b.x; o[5] = (__bf16)b.y; o[6] = (__bf16)b.z; o[7] = (__bf16)b.w;
    *(v8bf*)(d + i) = o;
}

// ---------------------------------------------------------------------------
__global__ void mask_init_kernel(int* __restrict__ mask) {
    int e = blockIdx.x * blockDim.x + threadIdx.x;
    if (e < BB * LL) mask[e] = 0;
}

// gather selected sentence rows (zeros at step 0): ssel (B x 1024 f32)
__global__ void gather_kernel(const float* __restrict__ sent,
                              const int* __restrict__ idxbuf,
                              float* __restrict__ ssel, int step)
{
    const int b = blockIdx.x;
    const int t = threadIdx.x;           // 256 threads, 4 floats each
    float4 v = make_float4(0.f, 0.f, 0.f, 0.f);
    if (step > 0) {
        const int id = idxbuf[b];
        v = *(const float4*)(sent + ((size_t)b * LL + id) * D2 + t * 4);
    }
    *(float4*)(ssel + (size_t)b * D2 + t * 4) = v;
}

// GRU elementwise combine: h = (1-z)*n + z*h over (B x EH)
__global__ void gru_combine_kernel(const float* __restrict__ gx,
                                   const float* __restrict__ gh,
                                   float* __restrict__ h)
{
    const int e = blockIdx.x * blockDim.x + threadIdx.x;   // B*EH = 65536
    const int b = e >> 9;
    const int j = e & 511;
    const float* gxb = gx + (size_t)b * G3;
    const float* ghb = gh + (size_t)b * G3;
    const float r = sigf(gxb[j]         + ghb[j]);
    const float z = sigf(gxb[EH + j]    + ghb[EH + j]);
    const float n = tanhf(gxb[2*EH + j] + r * ghb[2*EH + j]);
    h[e] = (1.0f - z) * n + z * h[e];
}

// score dot: sc[b][l] = ws . tanh(q[b] + d_proj[b][l]) + bs ; one block per (b,l)
__global__ __launch_bounds__(128) void score_kernel(
    const float* __restrict__ q,
    const float* __restrict__ dproj,
    const float* __restrict__ wsel,      // (A,)
    const float* __restrict__ bs,        // (1,)
    float* __restrict__ sctmp)           // (B*L,)
{
    __shared__ float red[128];
    const int bl = blockIdx.x;           // b*L + l
    const int b  = bl / LL;
    const int t  = threadIdx.x;
    const int a0 = t * 4;                // 128 threads * 4 = 512

    const float4 qv = *(const float4*)(q + (size_t)b * AA + a0);
    const float4 dv = *(const float4*)(dproj + (size_t)bl * AA + a0);
    const float4 wv = *(const float4*)(wsel + a0);

    float p = wv.x * tanhf(qv.x + dv.x)
            + wv.y * tanhf(qv.y + dv.y)
            + wv.z * tanhf(qv.z + dv.z)
            + wv.w * tanhf(qv.w + dv.w);

    red[t] = p;
    __syncthreads();
#pragma unroll
    for (int s = 64; s > 0; s >>= 1) {
        if (t < s) red[t] += red[t + s];
        __syncthreads();
    }
    if (t == 0) sctmp[bl] = red[0] + bs[0];
}

// masked argmax per batch row; writes masked scores + selected index to d_out
__global__ __launch_bounds__(256) void argmax_kernel(
    const float* __restrict__ sctmp,
    int* __restrict__ mask,
    int* __restrict__ idxbuf,
    float* __restrict__ out, int step)
{
    __shared__ float sv[256];
    __shared__ int   si[256];
    const int b = blockIdx.x;
    const int t = threadIdx.x;

    float v = -3.0e38f;
    int   li = t;
    if (t < LL) {
        float sc = sctmp[b * LL + t];
        if (mask[b * LL + t]) sc = -1000000.0f;
        out[(size_t)b * TT * LL + (size_t)step * LL + t] = sc;
        v = sc;
    }
    sv[t] = v; si[t] = li;
    __syncthreads();
#pragma unroll
    for (int s = 128; s > 0; s >>= 1) {
        if (t < s) {
            // prefer larger value; on tie, lower index (jnp.argmax semantics)
            if (sv[t + s] > sv[t] || (sv[t + s] == sv[t] && si[t + s] < si[t])) {
                sv[t] = sv[t + s];
                si[t] = si[t + s];
            }
        }
        __syncthreads();
    }
    if (t == 0) {
        const int id = si[0];
        idxbuf[b] = id;
        mask[b * LL + id] = 1;
        out[(size_t)BB * TT * LL + (size_t)b * TT + step] = (float)id;
    }
}

// ---------------------------------------------------------------------------
extern "C" void kernel_launch(void* const* d_in, const int* in_sizes, int n_in,
                              void* d_out, int out_size, void* d_ws, size_t ws_size,
                              hipStream_t stream)
{
    const float* sent     = (const float*)d_in[0];   // (B, L, 2*SH)
    const float* h0_w     = (const float*)d_in[1];   // (EH, SH)
    const float* h0_b     = (const float*)d_in[2];   // (EH,)
    const float* gru_w_ih = (const float*)d_in[3];   // (3EH, 2SH)
    const float* gru_w_hh = (const float*)d_in[4];   // (3EH, EH)
    const float* gru_b_ih = (const float*)d_in[5];   // (3EH,)
    const float* gru_b_hh = (const float*)d_in[6];   // (3EH,)
    const float* wq       = (const float*)d_in[7];   // (A, EH)
    const float* bq       = (const float*)d_in[8];   // (A,)
    const float* wd       = (const float*)d_in[9];   // (A, 2SH)
    const float* bd       = (const float*)d_in[10];  // (A,)
    const float* wsel     = (const float*)d_in[11];  // (1, A)
    const float* bs       = (const float*)d_in[12];  // (1,)

    float* out = (float*)d_out;

    // ---------------- workspace layout ----------------
    float* w = (float*)d_ws;
    float* dproj = w;                   w += (size_t)BB * LL * AA;   // 13,107,200
    float* h     = w;                   w += (size_t)BB * EH;
    float* q     = w;                   w += (size_t)BB * AA;
    float* gx    = w;                   w += (size_t)BB * G3;
    float* gh    = w;                   w += (size_t)BB * G3;
    float* ssel  = w;                   w += (size_t)BB * D2;
    float* sctmp = w;                   w += (size_t)BB * LL;
    int*   mask  = (int*)w;             w += (size_t)BB * LL;
    int*   idxb  = (int*)w;             w += BB;
    // bf16 weight buffers (region starts 16B-aligned; all sizes multiples of 8)
    __bf16* bw = (__bf16*)w;
    __bf16* h0wbf = bw;                 bw += (size_t)EH * SH;
    __bf16* wdbf  = bw;                 bw += (size_t)AA * D2;
    __bf16* wihbf = bw;                 bw += (size_t)G3 * D2;
    __bf16* whhbf = bw;                 bw += (size_t)G3 * EH;
    __bf16* wqbf  = bw;                 bw += (size_t)AA * EH;

    const dim3 gblk(128);
    const dim3 cblk(256);

    // ---- pre-pass: mask = 0, weights f32 -> bf16 (amortized across reuse) ----
    mask_init_kernel<<<dim3((BB * LL + 255) / 256), cblk, 0, stream>>>(mask);
    cvt_bf16_kernel<<<dim3((EH * SH / 8 + 255) / 256), cblk, 0, stream>>>(h0_w,     h0wbf, EH * SH);
    cvt_bf16_kernel<<<dim3((AA * D2 / 8 + 255) / 256), cblk, 0, stream>>>(wd,       wdbf,  AA * D2);
    cvt_bf16_kernel<<<dim3((G3 * D2 / 8 + 255) / 256), cblk, 0, stream>>>(gru_w_ih, wihbf, G3 * D2);
    cvt_bf16_kernel<<<dim3((G3 * EH / 8 + 255) / 256), cblk, 0, stream>>>(gru_w_hh, whhbf, G3 * EH);
    cvt_bf16_kernel<<<dim3((AA * EH / 8 + 255) / 256), cblk, 0, stream>>>(wq,       wqbf,  AA * EH);

    // h0 = tanh(sent[:,0,SH:] @ h0_w^T + h0_b)  (X rows strided by L*D2, offset SH)
    wmma_gemm<<<dim3(EH / 64, BB / 64), gblk, 0, stream>>>(
        sent + SH, (long long)LL * D2, h0wbf, h0_b, h, EH, SH, /*act=*/1);

    // d_proj = sent @ wd^T + bd   (25600 x 512, K=1024) — the big GEMM
    wmma_gemm<<<dim3(AA / 64, (BB * LL) / 64), gblk, 0, stream>>>(
        sent, (long long)D2, wdbf, bd, dproj, AA, D2, /*act=*/0);

    for (int t = 0; t < TT; ++t) {
        // s_prev row per batch (zeros at t=0, else selected sentence)
        gather_kernel<<<dim3(BB), cblk, 0, stream>>>(sent, idxb, ssel, t);

        // gx = ssel @ W_ih^T + b_ih   (128 x 1536, K=1024)
        wmma_gemm<<<dim3(G3 / 64, BB / 64), gblk, 0, stream>>>(
            ssel, (long long)D2, wihbf, gru_b_ih, gx, G3, D2, 0);

        // gh = h @ W_hh^T + b_hh      (128 x 1536, K=512)
        wmma_gemm<<<dim3(G3 / 64, BB / 64), gblk, 0, stream>>>(
            h, (long long)EH, whhbf, gru_b_hh, gh, G3, EH, 0);

        // GRU combine -> new h
        gru_combine_kernel<<<dim3((BB * EH) / 256), cblk, 0, stream>>>(gx, gh, h);

        // q = h @ wq^T + bq           (128 x 512)
        wmma_gemm<<<dim3(AA / 64, BB / 64), gblk, 0, stream>>>(
            h, (long long)EH, wqbf, bq, q, AA, EH, 0);

        // scores for every (b, l)
        score_kernel<<<dim3(BB * LL), gblk, 0, stream>>>(q, dproj, wsel, bs, sctmp);

        // masked argmax + output write
        argmax_kernel<<<dim3(BB), cblk, 0, stream>>>(sctmp, mask, idxb, out, t);
    }
}
